// LinFE_74637941670203
// MI455X (gfx1250) — matrix operand order repaired and checked by reading, not compiled, not run on hardware
//
#include <hip/hip_runtime.h>
#include <hip/hip_bf16.h>

typedef _Float16 hv8  __attribute__((ext_vector_type(8)));
typedef _Float16 hv16 __attribute__((ext_vector_type(16)));
typedef float    fv8  __attribute__((ext_vector_type(8)));

#define TWO_PI_F 6.283185307179586f

__device__ __forceinline__ fv8 zero8() {
  fv8 z;
#pragma unroll
  for (int i = 0; i < 8; ++i) z[i] = 0.0f;
  return z;
}

#define CAT16(lo, hi) __builtin_shufflevector(lo, hi, 0,1,2,3,4,5,6,7,8,9,10,11,12,13,14,15)

// Positional-encode feature f (0..43) of a 16-feature vector with the 7 "main"
// periods [2,4,..,14]: layout [raw 16 | sin(x0,p0..6) cos(x0,p0..6) sin(x1,..) cos(x1,..)]
__device__ __forceinline__ float pe_main_feat(const float* __restrict__ x16, int f) {
  if (f < 16) return x16[f];
  int e = f - 16;
  int c = e / 14;           // which coord (0/1)
  int i = e % 14;           // 0..6 sin, 7..13 cos
  float per = 2.0f * (float)((i % 7) + 1);
  float ang = TWO_PI_F * x16[c] / per;
  return (i < 7) ? __sinf(ang) : __cosf(ang);
}

// D-tile writeback: lane holds column n, rows m = mbase + g*8 + r. Fused bias+relu.
__device__ __forceinline__ void store_tile(_Float16* actOut, fv8 acc, int mbase,
                                           int n, float bv, int g, int ldOut) {
#pragma unroll
  for (int r = 0; r < 8; ++r) {
    float v = acc[r] + bv;
    v = v > 0.0f ? v : 0.0f;
    actOut[(mbase + g * 8 + r) * ldOut + n] = (_Float16)v;
  }
}

// One fused dense layer: actOut[64,N] = relu(actIn[64,Kpad] * W^T + b)
// W: (N x Kpad) row-major f16 in global (L2-resident), bias f32.
// actIn/actOut: LDS f16 tiles, 64 rows, leading dims ldIn/ldOut.
// Mapping (B-reuse oriented): wave w owns N-tile pairs {2w,2w+1}+16i and all
// four M-tiles -> per k-step: 2 global B loads are amortized over 8 WMMAs, so
// global weight traffic is exactly 1x W per workgroup; A comes from LDS.
__device__ __forceinline__ void layer_wmma(
    const _Float16* __restrict__ W, const float* __restrict__ bias,
    const _Float16* actIn, _Float16* actOut,
    int Kpad, int N, int ldIn, int ldOut)
{
  const int tid  = (int)threadIdx.x;
  const int wave = tid >> 5;
  const int lane = tid & 31;
  const int g    = lane >> 4;   // lane half-group
  const int nl   = lane & 15;
  const int numNT = N >> 4;

  // A-matrix: per ISA 16-bit A layout, M = lane%16 for both halves;
  // lane group g covers K range [k0+16g, +16) (contiguous, matching B).
  const _Float16* arow = actIn + nl * ldIn;   // m-tile 0 row; tiles offset by 16*ldIn

  for (int nt = wave * 2; nt < numNT; nt += 16) {
    fv8 acc0a = zero8(), acc0b = zero8();
    fv8 acc1a = zero8(), acc1b = zero8();
    fv8 acc2a = zero8(), acc2b = zero8();
    fv8 acc3a = zero8(), acc3b = zero8();
    // B-matrix: lane = N column (lane%16), lane group selects K half;
    // 16 contiguous f16 from weight row (nt*16+nl).
    const _Float16* brow0 = W + (size_t)(nt * 16 + nl) * Kpad + g * 16;
    const _Float16* brow1 = brow0 + (size_t)16 * Kpad;
    for (int k0 = 0; k0 < Kpad; k0 += 32) {
      hv16 B0 = *(const hv16*)(brow0 + k0);
      hv16 B1 = *(const hv16*)(brow1 + k0);
      const _Float16* ak = arow + k0 + g * 16;
      hv8 a00 = *(const hv8*)(ak);
      hv8 a01 = *(const hv8*)(ak + 8);
      hv8 a10 = *(const hv8*)(ak + 16 * ldIn);
      hv8 a11 = *(const hv8*)(ak + 16 * ldIn + 8);
      hv8 a20 = *(const hv8*)(ak + 32 * ldIn);
      hv8 a21 = *(const hv8*)(ak + 32 * ldIn + 8);
      hv8 a30 = *(const hv8*)(ak + 48 * ldIn);
      hv8 a31 = *(const hv8*)(ak + 48 * ldIn + 8);
      hv16 A0 = CAT16(a00, a01);
      hv16 A1 = CAT16(a10, a11);
      hv16 A2 = CAT16(a20, a21);
      hv16 A3 = CAT16(a30, a31);
      acc0a = __builtin_amdgcn_wmma_f32_16x16x32_f16(false, A0, false, B0, (short)0, acc0a, false, false);
      acc0b = __builtin_amdgcn_wmma_f32_16x16x32_f16(false, A0, false, B1, (short)0, acc0b, false, false);
      acc1a = __builtin_amdgcn_wmma_f32_16x16x32_f16(false, A1, false, B0, (short)0, acc1a, false, false);
      acc1b = __builtin_amdgcn_wmma_f32_16x16x32_f16(false, A1, false, B1, (short)0, acc1b, false, false);
      acc2a = __builtin_amdgcn_wmma_f32_16x16x32_f16(false, A2, false, B0, (short)0, acc2a, false, false);
      acc2b = __builtin_amdgcn_wmma_f32_16x16x32_f16(false, A2, false, B1, (short)0, acc2b, false, false);
      acc3a = __builtin_amdgcn_wmma_f32_16x16x32_f16(false, A3, false, B0, (short)0, acc3a, false, false);
      acc3b = __builtin_amdgcn_wmma_f32_16x16x32_f16(false, A3, false, B1, (short)0, acc3b, false, false);
    }
    const int n0 = nt * 16 + nl;
    const int n1 = n0 + 16;
    const float bv0 = bias[n0];
    const float bv1 = bias[n1];
    store_tile(actOut, acc0a,  0, n0, bv0, g, ldOut);
    store_tile(actOut, acc0b,  0, n1, bv1, g, ldOut);
    store_tile(actOut, acc1a, 16, n0, bv0, g, ldOut);
    store_tile(actOut, acc1b, 16, n1, bv1, g, ldOut);
    store_tile(actOut, acc2a, 32, n0, bv0, g, ldOut);
    store_tile(actOut, acc2b, 32, n1, bv1, g, ldOut);
    store_tile(actOut, acc3a, 48, n0, bv0, g, ldOut);
    store_tile(actOut, acc3b, 48, n1, bv1, g, ldOut);
  }
}

// ---- weight conversion: f32 (N,K) -> f16 (N,Kpad) row-major, zero K-padding
__global__ void cvt_kernel(_Float16* __restrict__ dst, const float* __restrict__ src,
                           int N, int K, int Kpad) {
  int i = blockIdx.x * 256 + threadIdx.x;
  if (i >= N * Kpad) return;
  int n = i / Kpad, k = i - n * Kpad;
  dst[i] = (k < K) ? (_Float16)src[n * K + k] : (_Float16)0.0f;
}

// ---- AGV path: 1 block = 1 batch element = 64 AGV rows.
// rows: concat(main_enc(44), agv_enc(44)) padded to 96 -> 512 -> 1024 -> 512 -> 256 -> max over 64
__global__ void agv_kernel(const float* __restrict__ agvs,
                           const _Float16* __restrict__ wE, const float* __restrict__ bE,
                           const _Float16* __restrict__ w0, const float* __restrict__ b0,
                           const _Float16* __restrict__ w1, const float* __restrict__ b1,
                           const _Float16* __restrict__ w2, const float* __restrict__ b2,
                           float* __restrict__ out) {
  extern __shared__ _Float16 smem[];
  _Float16* bufA = smem;               // 64 x 1024
  _Float16* bufB = smem + 64 * 1024;   // 64 x 512
  __shared__ float menc[44];
  const int b = blockIdx.x;
  const int tid = (int)threadIdx.x;
  const float* xmain = agvs + (size_t)b * 1040;
  if (tid < 44) menc[tid] = pe_main_feat(xmain, tid);
  __syncthreads();
  for (int e = tid; e < 64 * 96; e += 256) {
    int a = e / 96, c = e - a * 96;
    float v;
    if (c < 44) v = menc[c];
    else if (c < 88) v = pe_main_feat(agvs + (size_t)b * 1040 + 16 + a * 16, c - 44);
    else v = 0.0f;
    bufA[e] = (_Float16)v;
  }
  __syncthreads();
  layer_wmma(wE, bE, bufA, bufB, 96, 512, 96, 512);     __syncthreads();
  layer_wmma(w0, b0, bufB, bufA, 512, 1024, 512, 1024); __syncthreads();
  layer_wmma(w1, b1, bufA, bufB, 1024, 512, 1024, 512); __syncthreads();
  layer_wmma(w2, b2, bufB, bufA, 512, 256, 512, 256);   __syncthreads();
  // max over the 64 AGV rows; one column per thread
  {
    int n = tid;
    float mx = (float)bufA[n];
    for (int r = 1; r < 64; ++r) {
      float v = (float)bufA[r * 256 + n];
      mx = v > mx ? v : mx;
    }
    out[(size_t)b * 768 + 256 + n] = mx;
  }
}

// ---- Station path: 1 block = 4 batch elements x 16 stations = 64 rows.
// rows: concat(main_enc(44), st_enc(20)) = 64 -> 512 -> 1024 -> 512 -> 256 -> max over 16
__global__ void station_kernel(const float* __restrict__ agvs,
                               const float* __restrict__ stations,
                               const _Float16* __restrict__ wE, const float* __restrict__ bE,
                               const _Float16* __restrict__ w0, const float* __restrict__ b0,
                               const _Float16* __restrict__ w1, const float* __restrict__ b1,
                               const _Float16* __restrict__ w2, const float* __restrict__ b2,
                               float* __restrict__ out) {
  extern __shared__ _Float16 smem[];
  _Float16* bufA = smem;               // 64 x 1024
  _Float16* bufB = smem + 64 * 1024;   // 64 x 512
  __shared__ float menc4[4][44];
  const int bq = blockIdx.x * 4;
  const int tid = (int)threadIdx.x;
  if (tid < 176) {
    int bi = tid / 44, f = tid - bi * 44;
    menc4[bi][f] = pe_main_feat(agvs + (size_t)(bq + bi) * 1040, f);
  }
  __syncthreads();
  for (int e = tid; e < 64 * 64; e += 256) {
    int r = e >> 6, c = e & 63;
    int bi = r >> 4, s = r & 15;
    float v;
    if (c < 44) v = menc4[bi][c];
    else {
      const float* xs = stations + (size_t)(bq + bi) * 256 + s * 16;
      int f = c - 44;
      if (f < 16) v = xs[f];
      else {            // period-1 encoding: [sin(x0),cos(x0),sin(x1),cos(x1)]
        int e2 = f - 16;
        float ang = TWO_PI_F * xs[e2 >> 1];
        v = (e2 & 1) ? __cosf(ang) : __sinf(ang);
      }
    }
    bufA[e] = (_Float16)v;
  }
  __syncthreads();
  layer_wmma(wE, bE, bufA, bufB, 64, 512, 64, 512);     __syncthreads();
  layer_wmma(w0, b0, bufB, bufA, 512, 1024, 512, 1024); __syncthreads();
  layer_wmma(w1, b1, bufA, bufB, 1024, 512, 1024, 512); __syncthreads();
  layer_wmma(w2, b2, bufB, bufA, 512, 256, 512, 256);   __syncthreads();
  for (int o = tid; o < 4 * 256; o += 256) {
    int bi = o >> 8, n = o & 255;
    float mx = (float)bufA[(bi * 16) * 256 + n];
    for (int r = 1; r < 16; ++r) {
      float v = (float)bufA[(bi * 16 + r) * 256 + n];
      mx = v > mx ? v : mx;
    }
    out[(size_t)(bq + bi) * 768 + 512 + n] = mx;
  }
}

// ---- Main path: 1 block = 64 batch rows.
// main_enc(44 pad 64) -> 256, concat last_action(64) = 320 -> 512 -> 512 -> 256
__global__ void main_kernel(const float* __restrict__ agvs,
                            const float* __restrict__ last_action,
                            const _Float16* __restrict__ wE, const float* __restrict__ bE,
                            const _Float16* __restrict__ w0, const float* __restrict__ b0,
                            const _Float16* __restrict__ w1, const float* __restrict__ b1,
                            const _Float16* __restrict__ w2, const float* __restrict__ b2,
                            float* __restrict__ out) {
  extern __shared__ _Float16 smem[];
  _Float16* bufA = smem;              // 64 x 512
  _Float16* bufB = smem + 64 * 512;   // 64 x 512 (first used as 64 x 320)
  const int row0 = blockIdx.x * 64;
  const int tid = (int)threadIdx.x;
  for (int e = tid; e < 64 * 64; e += 256) {
    int m = e >> 6, c = e & 63;
    float v = (c < 44) ? pe_main_feat(agvs + (size_t)(row0 + m) * 1040, c) : 0.0f;
    bufA[e] = (_Float16)v;
  }
  __syncthreads();
  layer_wmma(wE, bE, bufA, bufB, 64, 256, 64, 320);   // cols 0..255 of 320-wide buffer
  // stage last_action into cols 256..319 (disjoint region, no race with layer writes)
  for (int e = tid; e < 64 * 64; e += 256) {
    int m = e >> 6, c = e & 63;
    bufB[m * 320 + 256 + c] = (_Float16)last_action[(size_t)(row0 + m) * 64 + c];
  }
  __syncthreads();
  layer_wmma(w0, b0, bufB, bufA, 320, 512, 320, 512); __syncthreads();
  layer_wmma(w1, b1, bufA, bufB, 512, 512, 512, 512); __syncthreads();
  layer_wmma(w2, b2, bufB, bufA, 512, 256, 512, 256); __syncthreads();
  for (int o = tid; o < 64 * 256; o += 256) {
    int m = o >> 8, n = o & 255;
    out[(size_t)(row0 + m) * 768 + n] = (float)bufA[m * 256 + n];
  }
}

extern "C" void kernel_launch(void* const* d_in, const int* in_sizes, int n_in,
                              void* d_out, int out_size, void* d_ws, size_t ws_size,
                              hipStream_t stream) {
  (void)in_sizes; (void)n_in; (void)out_size; (void)ws_size;
  const float* agvs        = (const float*)d_in[0];
  const float* stations    = (const float*)d_in[1];
  const float* last_action = (const float*)d_in[2];
  _Float16* wsp = (_Float16*)d_ws;
  float* out = (float*)d_out;

  // f16 weight workspace offsets (in halves), K padded to multiple of 32
  const size_t OFF_EA = 0;
  const size_t OFF_EM = OFF_EA + (size_t)512 * 96;
  const size_t OFF_ES = OFF_EM + (size_t)256 * 64;
  const size_t OFF_A0 = OFF_ES + (size_t)512 * 64;
  const size_t OFF_A1 = OFF_A0 + (size_t)1024 * 512;
  const size_t OFF_A2 = OFF_A1 + (size_t)512 * 1024;
  const size_t OFF_S0 = OFF_A2 + (size_t)256 * 512;
  const size_t OFF_S1 = OFF_S0 + (size_t)1024 * 512;
  const size_t OFF_S2 = OFF_S1 + (size_t)512 * 1024;
  const size_t OFF_F0 = OFF_S2 + (size_t)256 * 512;
  const size_t OFF_F1 = OFF_F0 + (size_t)512 * 320;
  const size_t OFF_F2 = OFF_F1 + (size_t)512 * 512;

  // param flattening order (alphabetical dict keys, (W,b) pairs):
  // 3/4 embed_agvs, 5/6 embed_main, 7/8 embed_station,
  // 9..14 lint_agvs, 15..20 lint_station, 21..26 main_ff
  struct Cvt { int src; int N, K, Kp; size_t off; };
  const Cvt cv[12] = {
    {3,  512,   88,   96, OFF_EA}, {5,  256,   44,   64, OFF_EM},
    {7,  512,   64,   64, OFF_ES},
    {9, 1024,  512,  512, OFF_A0}, {11, 512, 1024, 1024, OFF_A1},
    {13, 256,  512,  512, OFF_A2},
    {15,1024,  512,  512, OFF_S0}, {17, 512, 1024, 1024, OFF_S1},
    {19, 256,  512,  512, OFF_S2},
    {21, 512,  320,  320, OFF_F0}, {23, 512,  512,  512, OFF_F1},
    {25, 256,  512,  512, OFF_F2},
  };
  for (int i = 0; i < 12; ++i) {
    int total = cv[i].N * cv[i].Kp;
    cvt_kernel<<<(total + 255) / 256, 256, 0, stream>>>(
        wsp + cv[i].off, (const float*)d_in[cv[i].src], cv[i].N, cv[i].K, cv[i].Kp);
  }

  const int LDS_BIG  = (64 * 1024 + 64 * 512) * 2;  // 196608 B
  const int LDS_MAIN = (64 * 512 + 64 * 512) * 2;   // 131072 B
  (void)hipFuncSetAttribute((const void*)agv_kernel,
        hipFuncAttributeMaxDynamicSharedMemorySize, LDS_BIG);
  (void)hipFuncSetAttribute((const void*)station_kernel,
        hipFuncAttributeMaxDynamicSharedMemorySize, LDS_BIG);
  (void)hipFuncSetAttribute((const void*)main_kernel,
        hipFuncAttributeMaxDynamicSharedMemorySize, LDS_MAIN);

  agv_kernel<<<1024, 256, LDS_BIG, stream>>>(agvs,
      wsp + OFF_EA, (const float*)d_in[4],
      wsp + OFF_A0, (const float*)d_in[10],
      wsp + OFF_A1, (const float*)d_in[12],
      wsp + OFF_A2, (const float*)d_in[14], out);

  station_kernel<<<256, 256, LDS_BIG, stream>>>(agvs, stations,
      wsp + OFF_ES, (const float*)d_in[8],
      wsp + OFF_S0, (const float*)d_in[16],
      wsp + OFF_S1, (const float*)d_in[18],
      wsp + OFF_S2, (const float*)d_in[20], out);

  main_kernel<<<16, 256, LDS_MAIN, stream>>>(agvs, last_action,
      wsp + OFF_EM, (const float*)d_in[6],
      wsp + OFF_F0, (const float*)d_in[22],
      wsp + OFF_F1, (const float*)d_in[24],
      wsp + OFF_F2, (const float*)d_in[26], out);
}
